// BBoxVAE_2138893714161
// MI455X (gfx1250) — matrix operand order repaired and checked by reading, not compiled, not run on hardware
//
#include <hip/hip_runtime.h>
#include <hip/hip_bf16.h>

// ---------------------------------------------------------------------------
// BBoxVAE forward on MI455X (gfx1250, wave32, WMMA).
// Block = 256 threads (8 waves) = 64 batch rows (4 WMMA M-tiles).
// Each B-fragment load is reused across 4 M-tiles -> 4x less L2 weight traffic
// than a 16-row block. All GEMMs -> v_wmma_f32_16x16x32_bf16 (f32 accum).
// Weights pre-packed to B-fragment layout in d_ws (bf16, ~470KB, L2 resident).
// ---------------------------------------------------------------------------

typedef __bf16 bf16_t;
typedef __attribute__((ext_vector_type(16))) __bf16 v16bf;
typedef __attribute__((ext_vector_type(8)))  __bf16 v8bf;
typedef __attribute__((ext_vector_type(8)))  float  v8f;

#define BATCH 32768
#define SEQL  24
#define ROWS  64           // batch rows per block (4 M-tiles)

// ---- packed weight offsets (bf16 elements), fragment = [32 lanes][16 bf16] --
constexpr int SZ_CW1 = 32*128,  OFF_CW1 = 0;
constexpr int SZ_CW2 = 128*128, OFF_CW2 = OFF_CW1 + SZ_CW1;
constexpr int SZ_CW5 = 128*128, OFF_CW5 = OFF_CW2 + SZ_CW2;
constexpr int SZ_CWO = 384*128, OFF_CWO = OFF_CW5 + SZ_CW5;
constexpr int SZ_PW1 = 128*32,  OFF_PW1 = OFF_CWO + SZ_CWO;
constexpr int SZ_PWM = 32*32,   OFF_PWM = OFF_PW1 + SZ_PW1;
constexpr int SZ_PWV = 32*32,   OFF_PWV = OFF_PWM + SZ_PWM;
constexpr int SZ_EW1 = 32*128,  OFF_EW1 = OFF_PWV + SZ_PWV;
constexpr int SZ_EW2 = 128*128, OFF_EW2 = OFF_EW1 + SZ_EW1;
constexpr int SZ_EW3 = 256*32,  OFF_EW3 = OFF_EW2 + SZ_EW2;
constexpr int SZ_EWM = 32*32,   OFF_EWM = OFF_EW3 + SZ_EW3;
constexpr int SZ_EWV = 32*32,   OFF_EWV = OFF_EWM + SZ_EWM;
constexpr int SZ_DW1 = 160*128, OFF_DW1 = OFF_EWV + SZ_EWV;
constexpr int SZ_DW2 = 128*64,  OFF_DW2 = OFF_DW1 + SZ_DW1;
constexpr int SZ_DW3 = 64*16,   OFF_DW3 = OFF_DW2 + SZ_DW2;
constexpr int SZ_LK  = 32*512,  OFF_LK  = OFF_DW3 + SZ_DW3;
constexpr int SZ_LR  = 128*512, OFF_LR  = OFF_LK + SZ_LK;
constexpr int TOT_PACK = OFF_LR + SZ_LR;   // 234496 bf16 = 468992 B

// ---------------------------------------------------------------------------
__device__ __forceinline__ v8f wmma_bf16(v16bf a, v16bf b, v8f c) {
  return __builtin_amdgcn_wmma_f32_16x16x32_bf16(false, a, false, b, (short)0, c,
                                                 false, false);
}

// A fragment (16x32 bf16) from LDS row-major [16][ldK]:
// lane -> row M=lane%16; lanes 0-15 hold K {0..7,16..23}, lanes 16-31 hold
// K {8..15,24..31} -> two contiguous 16B LDS loads.
__device__ __forceinline__ v16bf load_afrag(const bf16_t* base, int ldK) {
  const int lane = threadIdx.x & 31;
  const bf16_t* p = base + (lane & 15) * ldK + ((lane >> 4) << 3);
  v8bf lo = *(const v8bf*)p;
  v8bf hi = *(const v8bf*)(p + 16);
  v16bf a;
#pragma unroll
  for (int i = 0; i < 8; ++i) { a[i] = lo[i]; a[i + 8] = hi[i]; }
  return a;
}

// B fragment (32x16 bf16) from packed global: 32B contiguous per lane.
__device__ __forceinline__ v16bf load_bfrag(const bf16_t* wp, int frag) {
  const int lane = threadIdx.x & 31;
  return *(const v16bf*)(wp + ((size_t)frag * 32 + lane) * 16);
}

__device__ __forceinline__ float sigm(float x) { return 1.f / (1.f + __expf(-x)); }

// Tiled GEMM: [64,K] x [K,16] for one N-tile `nt`, 4 M-tiles per B-frag load.
// A split in two segments (for concat inputs). C layout: VGPR r, lane l ->
// row = r + 8*(l>>4), col = l&15.
template<int KT0, int KT1>
__device__ __forceinline__ void gemm4(
    const bf16_t* __restrict__ A0, int ldA0, int kOff0,
    const bf16_t* __restrict__ A1, int ldA1, int kOff1,
    const bf16_t* __restrict__ wp, int wNT, int nt,
    const float* __restrict__ bias, int relu_act,
    bf16_t* __restrict__ out0, int ld0,
    float* __restrict__ outF, int ldF)
{
  v8f acc[4] = {{}, {}, {}, {}};
#pragma unroll
  for (int kt = 0; kt < KT0; ++kt) {
    v16bf b = load_bfrag(wp, (kOff0 + kt) * wNT + nt);
#pragma unroll
    for (int m = 0; m < 4; ++m) {
      v16bf a = load_afrag(A0 + (m * 16) * ldA0 + kt * 32, ldA0);
      acc[m] = wmma_bf16(a, b, acc[m]);
    }
  }
  if constexpr (KT1 > 0) {
#pragma unroll
    for (int kt = 0; kt < KT1; ++kt) {
      v16bf b = load_bfrag(wp, (kOff1 + kt) * wNT + nt);
#pragma unroll
      for (int m = 0; m < 4; ++m) {
        v16bf a = load_afrag(A1 + (m * 16) * ldA1 + kt * 32, ldA1);
        acc[m] = wmma_bf16(a, b, acc[m]);
      }
    }
  }
  const int lane = threadIdx.x & 31;
  const int col = lane & 15, rb = (lane >> 4) * 8;
  const float bv = bias[nt * 16 + col];
#pragma unroll
  for (int m = 0; m < 4; ++m) {
#pragma unroll
    for (int r = 0; r < 8; ++r) {
      float v = acc[m][r] + bv;
      if (relu_act) v = fmaxf(v, 0.f);
      const int row = m * 16 + rb + r;
      if (out0) out0[row * ld0 + nt * 16 + col] = (bf16_t)v;
      if (outF) outF[row * ldF + nt * 16 + col] = v;
    }
  }
}

// ---------------------------------------------------------------------------
// Pre-pass kernels
// ---------------------------------------------------------------------------
__global__ void pack_weight(const float* __restrict__ W, bf16_t* __restrict__ out,
                            int Kr, int Nr, int Np) {
  const int lane = threadIdx.x;          // 32 threads
  const int frag = blockIdx.x;
  const int ntiles = Np >> 4;
  const int kt = frag / ntiles, nt = frag - kt * ntiles;
  const int col = nt * 16 + (lane & 15);
  const int kb  = kt * 32 + (lane >> 4) * 8;
  bf16_t* o = out + ((size_t)frag * 32 + lane) * 16;
#pragma unroll
  for (int i = 0; i < 16; ++i) {
    int k = kb + (i & 7) + ((i >> 3) << 4);
    float v = (k < Kr && col < Nr) ? W[(size_t)k * Nr + col] : 0.f;
    o[i] = (bf16_t)v;
  }
}

// i2_all[t][j] = relu( relu(cw3[t,:]+cb3) @ cw4 + cb4 )
__global__ void compute_i2(const float* __restrict__ cw3, const float* __restrict__ cb3,
                           const float* __restrict__ cw4, const float* __restrict__ cb4,
                           float* __restrict__ i2all) {
  const int t = blockIdx.x, j = threadIdx.x;   // 24 x 128
  float acc = cb4[j];
  for (int k = 0; k < 128; ++k) {
    float hk = fmaxf(cw3[t * 128 + k] + cb3[k], 0.f);
    acc += hk * cw4[k * 128 + j];
  }
  i2all[t * 128 + j] = fmaxf(acc, 0.f);
}

// i2cond[t][j] = cbo[j] + i2_all[t,:] @ cwo[128:256, j]   (batch-invariant part
// of the cond GEMM, folded in as per-step bias)
__global__ void compute_i2cond(const float* __restrict__ i2all,
                               const float* __restrict__ cwo,
                               const float* __restrict__ cbo,
                               float* __restrict__ i2cond) {
  const int t = blockIdx.x, j = threadIdx.x;   // 24 x 128
  float acc = cbo[j];
  for (int k = 0; k < 128; ++k)
    acc += i2all[t * 128 + k] * cwo[(size_t)(128 + k) * 128 + j];
  i2cond[t * 128 + j] = acc;
}

// ---------------------------------------------------------------------------
// Main kernel: 256 threads (8 waves) per 64 batch rows.
// ---------------------------------------------------------------------------
__global__ void __launch_bounds__(256)
vae_main(const float* __restrict__ label_set,
         const float* __restrict__ bbox_input,
         const float* __restrict__ eps,
         const float* __restrict__ i2cond,
         const bf16_t* __restrict__ wpk,
         const float* __restrict__ cb1, const float* __restrict__ cb2,
         const float* __restrict__ cb5,
         const float* __restrict__ pb1, const float* __restrict__ pbm,
         const float* __restrict__ pbv,
         const float* __restrict__ eb1, const float* __restrict__ eb2,
         const float* __restrict__ eb3, const float* __restrict__ ebm,
         const float* __restrict__ ebv,
         const float* __restrict__ db1, const float* __restrict__ db2,
         const float* __restrict__ db3,
         const float* __restrict__ lb,
         float* __restrict__ out_bbox, float* __restrict__ out_kl)
{
  __shared__ __align__(32) bf16_t sI1[ROWS * 128];     // i1 (static after init)
  __shared__ __align__(32) bf16_t sI3[ROWS * 128];     // i3
  __shared__ __align__(32) bf16_t sE1[ROWS * 128];     // e1 ; aliased: sGF
  __shared__ __align__(32) bf16_t sT128[ROWS * 128];   // enc1/d1 ; aliased: sGI
  __shared__ __align__(32) bf16_t sD[ROWS * 160];      // [cond | z]
  __shared__ __align__(32) bf16_t sT64[ROWS * 64];     // d2
  __shared__ __align__(32) bf16_t sX[ROWS * 32];       // [onehot24|bbox4|0]
  __shared__ __align__(32) bf16_t sGT[ROWS * 32];      // gt padded / label stage
  __shared__ __align__(32) bf16_t sH[2][ROWS * 128];   // LSTM h double-buffered
  __shared__ __align__(32) bf16_t sPH[ROWS * 32];
  __shared__ __align__(32) bf16_t sINTER[ROWS * 32];
  __shared__ float sC[ROWS * 128];                     // LSTM cell (f32)
  __shared__ float sZMC[ROWS * 32], sZVC[ROWS * 32], sZM[ROWS * 32], sZV[ROWS * 32];

  bf16_t* sGI = sT128;   // sigmoid(i-gate), pass1 -> pass2 (disjoint lifetime)
  bf16_t* sGF = sE1;     // sigmoid(f-gate)

  const int tid  = threadIdx.x;
  const int lane = tid & 31;
  const int wave = tid >> 5;
  const int r0   = blockIdx.x * ROWS;

  // ---- init: zero h/c, stage labels, i1 = relu(relu(label@cw1+cb1)@cw2+cb2)
  for (int q = tid; q < ROWS * 128; q += 256) { sH[0][q] = (bf16_t)0.f; sC[q] = 0.f; }
  for (int q = tid; q < ROWS * 32; q += 256) {
    int r = q >> 5, j = q & 31;
    float v = (j < 24) ? label_set[(size_t)(r0 + r) * 24 + j] : 0.f;
    sGT[q] = (bf16_t)v;
  }
  __syncthreads();
  gemm4<1, 0>(sGT, 32, 0, nullptr, 0, 0, wpk + OFF_CW1, 8, wave, cb1, 1,
              sT128, 128, nullptr, 0);
  __syncthreads();
  gemm4<4, 0>(sT128, 128, 0, nullptr, 0, 0, wpk + OFF_CW2, 8, wave, cb2, 1,
              sI1, 128, nullptr, 0);
  __syncthreads();

  int par = 0;
  for (int t = 0; t < SEQL; ++t) {
    // ---- P0: per-step fills (gt tile, LSTM one-hot)
    for (int q = tid; q < ROWS * 32; q += 256) {
      int r = q >> 5, j = q & 31;
      float gv = (j < 4) ? bbox_input[((size_t)(r0 + r) * SEQL + t) * 4 + j] : 0.f;
      sGT[q] = (bf16_t)gv;
      if (j < 24)       sX[q] = (bf16_t)((j == t) ? 1.f : 0.f);
      else if (j >= 28) sX[q] = (bf16_t)0.f;   // cols 24..27 filled by decoder
    }
    __syncthreads();

    // ---- P1: i3 = relu(h @ cw5 + cb5)
    gemm4<4, 0>(sH[par], 128, 0, nullptr, 0, 0, wpk + OFF_CW5, 8, wave, cb5, 1,
                sI3, 128, nullptr, 0);
    __syncthreads();

    // ---- P2: cond = i1@cwo[0:128] + i3@cwo[256:384] + i2cond[t] (bias)
    gemm4<4, 4>(sI1, 128, 0, sI3, 128, 8, wpk + OFF_CWO, 8, wave,
                i2cond + t * 128, 0, sD, 160, nullptr, 0);
    __syncthreads();

    // ---- P3: ph = relu(cond @ pw1 + pb1)
    if (wave < 2)
      gemm4<4, 0>(sD, 160, 0, nullptr, 0, 0, wpk + OFF_PW1, 2, wave, pb1, 1,
                  sPH, 32, nullptr, 0);
    __syncthreads();

    // ---- P4: prior heads zmc/zvc (waves 0-3) | enc1 (waves 4-7)
    if (wave < 4) {
      const int hsel = wave >> 1;
      gemm4<1, 0>(sPH, 32, 0, nullptr, 0, 0,
                  wpk + (hsel ? OFF_PWV : OFF_PWM), 2, wave & 1,
                  hsel ? pbv : pbm, 0, nullptr, 0, hsel ? sZVC : sZMC, 32);
    } else {
      for (int nt = wave - 4; nt < 8; nt += 4)
        gemm4<1, 0>(sGT, 32, 0, nullptr, 0, 0, wpk + OFF_EW1, 8, nt, eb1, 1,
                    sT128, 128, nullptr, 0);
    }
    __syncthreads();

    // ---- P5: e1 = enc1 @ ew2 + eb2 (no relu)
    gemm4<4, 0>(sT128, 128, 0, nullptr, 0, 0, wpk + OFF_EW2, 8, wave, eb2, 0,
                sE1, 128, nullptr, 0);
    __syncthreads();

    // ---- P6: inter = relu([e1 | cond] @ ew3 + eb3)
    if (wave < 2)
      gemm4<4, 4>(sE1, 128, 0, sD, 160, 4, wpk + OFF_EW3, 2, wave, eb3, 1,
                  sINTER, 32, nullptr, 0);
    __syncthreads();

    // ---- P7: encoder heads zm/zv
    if (wave < 4) {
      const int hsel = wave >> 1;
      gemm4<1, 0>(sINTER, 32, 0, nullptr, 0, 0,
                  wpk + (hsel ? OFF_EWV : OFF_EWM), 2, wave & 1,
                  hsel ? ebv : ebm, 0, nullptr, 0, hsel ? sZV : sZM, 32);
    }
    __syncthreads();

    // ---- P8: kl output + reparameterized z
    for (int q = tid; q < ROWS * 32; q += 256) {
      int r = q >> 5, j = q & 31;
      float zm = sZM[q], zv = sZV[q], zmc = sZMC[q], zvc = sZVC[q];
      float dm = zm - zmc;
      float kl = 0.5f * (zvc - zv + (__expf(zv) + dm * dm) * __expf(-zvc) - 1.f);
      size_t bi = (size_t)(r0 + r) * SEQL + t;
      out_kl[bi * 32 + j] = kl;
      float z = zm + __expf(0.5f * zv) * eps[bi * 32 + j];
      sD[r * 160 + 128 + j] = (bf16_t)z;
    }
    __syncthreads();

    // ---- P9/P10: decoder
    gemm4<5, 0>(sD, 160, 0, nullptr, 0, 0, wpk + OFF_DW1, 8, wave, db1, 1,
                sT128, 128, nullptr, 0);
    __syncthreads();
    if (wave < 4)
      gemm4<4, 0>(sT128, 128, 0, nullptr, 0, 0, wpk + OFF_DW2, 4, wave, db2, 1,
                  sT64, 64, nullptr, 0);
    __syncthreads();

    // ---- P11: bbox = sigmoid(d2 @ dw3 + db3) (wave 0, 4 M-tiles)
    if (wave == 0) {
      v8f acc[4] = {{}, {}, {}, {}};
#pragma unroll
      for (int kt = 0; kt < 2; ++kt) {
        v16bf b = load_bfrag(wpk + OFF_DW3, kt);   // wNT = 1
#pragma unroll
        for (int m = 0; m < 4; ++m) {
          v16bf a = load_afrag(sT64 + (m * 16) * 64 + kt * 32, 64);
          acc[m] = wmma_bf16(a, b, acc[m]);
        }
      }
      const int col = lane & 15, rb = (lane >> 4) * 8;
      if (col < 4) {
        float bv = db3[col];
#pragma unroll
        for (int m = 0; m < 4; ++m)
#pragma unroll
          for (int r = 0; r < 8; ++r) {
            const int row = m * 16 + rb + r;
            float v = sigm(acc[m][r] + bv);
            out_bbox[((size_t)(r0 + row) * SEQL + t) * 4 + col] = v;
            sX[row * 32 + 24 + col] = (bf16_t)v;
          }
      }
    }
    __syncthreads();

    // ---- P12: LSTM gates, column group g = wave, two passes (VGPR budget)
    {
      const int g = wave;
      const int col = lane & 15, rb = (lane >> 4) * 8;
      const int nc = g * 16 + col;
      // pass 1: i,f gates -> sigmoid -> LDS
      {
        v8f ai[4] = {{}, {}, {}, {}}, af[4] = {{}, {}, {}, {}};
        {
          v16bf bi = load_bfrag(wpk + OFF_LK, g);
          v16bf bf = load_bfrag(wpk + OFF_LK, g + 8);
#pragma unroll
          for (int m = 0; m < 4; ++m) {
            v16bf a = load_afrag(sX + (m * 16) * 32, 32);
            ai[m] = wmma_bf16(a, bi, ai[m]);
            af[m] = wmma_bf16(a, bf, af[m]);
          }
        }
#pragma unroll
        for (int kt = 0; kt < 4; ++kt) {
          v16bf bi = load_bfrag(wpk + OFF_LR, kt * 32 + g);
          v16bf bf = load_bfrag(wpk + OFF_LR, kt * 32 + g + 8);
#pragma unroll
          for (int m = 0; m < 4; ++m) {
            v16bf a = load_afrag(sH[par] + (m * 16) * 128 + kt * 32, 128);
            ai[m] = wmma_bf16(a, bi, ai[m]);
            af[m] = wmma_bf16(a, bf, af[m]);
          }
        }
        const float bI = lb[nc], bF = lb[128 + nc];
#pragma unroll
        for (int m = 0; m < 4; ++m)
#pragma unroll
          for (int r = 0; r < 8; ++r) {
            const int row = m * 16 + rb + r;
            sGI[row * 128 + nc] = (bf16_t)sigm(ai[m][r] + bI);
            sGF[row * 128 + nc] = (bf16_t)sigm(af[m][r] + bF);
          }
      }
      // pass 2: g,o gates + cell update (same wave owns same columns -> no bar)
      {
        v8f ag[4] = {{}, {}, {}, {}}, ao[4] = {{}, {}, {}, {}};
        {
          v16bf bg = load_bfrag(wpk + OFF_LK, g + 16);
          v16bf bo = load_bfrag(wpk + OFF_LK, g + 24);
#pragma unroll
          for (int m = 0; m < 4; ++m) {
            v16bf a = load_afrag(sX + (m * 16) * 32, 32);
            ag[m] = wmma_bf16(a, bg, ag[m]);
            ao[m] = wmma_bf16(a, bo, ao[m]);
          }
        }
#pragma unroll
        for (int kt = 0; kt < 4; ++kt) {
          v16bf bg = load_bfrag(wpk + OFF_LR, kt * 32 + g + 16);
          v16bf bo = load_bfrag(wpk + OFF_LR, kt * 32 + g + 24);
#pragma unroll
          for (int m = 0; m < 4; ++m) {
            v16bf a = load_afrag(sH[par] + (m * 16) * 128 + kt * 32, 128);
            ag[m] = wmma_bf16(a, bg, ag[m]);
            ao[m] = wmma_bf16(a, bo, ao[m]);
          }
        }
        const float bG = lb[256 + nc], bO = lb[384 + nc];
#pragma unroll
        for (int m = 0; m < 4; ++m)
#pragma unroll
          for (int r = 0; r < 8; ++r) {
            const int row = m * 16 + rb + r;
            const int idx = row * 128 + nc;
            float G = tanhf(ag[m][r] + bG);
            float O = sigm(ao[m][r] + bO);
            float I = (float)sGI[idx];
            float F = (float)sGF[idx];
            float cn = F * sC[idx] + I * G;
            sC[idx] = cn;
            sH[par ^ 1][idx] = (bf16_t)(O * tanhf(cn));
          }
      }
    }
    __syncthreads();
    par ^= 1;
  }
}

// ---------------------------------------------------------------------------
extern "C" void kernel_launch(void* const* d_in, const int* in_sizes, int n_in,
                              void* d_out, int out_size, void* d_ws, size_t ws_size,
                              hipStream_t stream) {
  (void)in_sizes; (void)n_in; (void)out_size; (void)ws_size;
  const float* label_set  = (const float*)d_in[0];
  const float* bbox_input = (const float*)d_in[1];
  const float* eps        = (const float*)d_in[2];
  const float* cw1 = (const float*)d_in[3];   const float* cb1 = (const float*)d_in[4];
  const float* cw2 = (const float*)d_in[5];   const float* cb2 = (const float*)d_in[6];
  const float* cw3 = (const float*)d_in[7];   const float* cb3 = (const float*)d_in[8];
  const float* cw4 = (const float*)d_in[9];   const float* cb4 = (const float*)d_in[10];
  const float* cw5 = (const float*)d_in[11];  const float* cb5 = (const float*)d_in[12];
  const float* cwo = (const float*)d_in[13];  const float* cbo = (const float*)d_in[14];
  const float* pw1 = (const float*)d_in[15];  const float* pb1 = (const float*)d_in[16];
  const float* pwm = (const float*)d_in[17];  const float* pbm = (const float*)d_in[18];
  const float* pwv = (const float*)d_in[19];  const float* pbv = (const float*)d_in[20];
  const float* ew1 = (const float*)d_in[21];  const float* eb1 = (const float*)d_in[22];
  const float* ew2 = (const float*)d_in[23];  const float* eb2 = (const float*)d_in[24];
  const float* ew3 = (const float*)d_in[25];  const float* eb3 = (const float*)d_in[26];
  const float* ewm = (const float*)d_in[27];  const float* ebm = (const float*)d_in[28];
  const float* ewv = (const float*)d_in[29];  const float* ebv = (const float*)d_in[30];
  const float* dw1 = (const float*)d_in[31];  const float* db1 = (const float*)d_in[32];
  const float* dw2 = (const float*)d_in[33];  const float* db2 = (const float*)d_in[34];
  const float* dw3 = (const float*)d_in[35];  const float* db3 = (const float*)d_in[36];
  const float* lk  = (const float*)d_in[37];
  const float* lr  = (const float*)d_in[38];
  const float* lb  = (const float*)d_in[39];

  bf16_t* wpk   = (bf16_t*)d_ws;
  float* i2all  = (float*)((char*)d_ws + (size_t)TOT_PACK * sizeof(bf16_t));
  float* i2cond = i2all + 24 * 128;

  auto pack = [&](const float* W, int off, int Kr, int Nr, int Kp, int Np) {
    dim3 g((Kp / 32) * (Np / 16));
    pack_weight<<<g, 32, 0, stream>>>(W, wpk + off, Kr, Nr, Np);
  };
  pack(cw1, OFF_CW1,  24, 128,  32, 128);
  pack(cw2, OFF_CW2, 128, 128, 128, 128);
  pack(cw5, OFF_CW5, 128, 128, 128, 128);
  pack(cwo, OFF_CWO, 384, 128, 384, 128);
  pack(pw1, OFF_PW1, 128,  32, 128,  32);
  pack(pwm, OFF_PWM,  32,  32,  32,  32);
  pack(pwv, OFF_PWV,  32,  32,  32,  32);
  pack(ew1, OFF_EW1,   4, 128,  32, 128);
  pack(ew2, OFF_EW2, 128, 128, 128, 128);
  pack(ew3, OFF_EW3, 256,  32, 256,  32);
  pack(ewm, OFF_EWM,  32,  32,  32,  32);
  pack(ewv, OFF_EWV,  32,  32,  32,  32);
  pack(dw1, OFF_DW1, 160, 128, 160, 128);
  pack(dw2, OFF_DW2, 128,  64, 128,  64);
  pack(dw3, OFF_DW3,  64,   4,  64,  16);
  pack(lk,  OFF_LK,   28, 512,  32, 512);
  pack(lr,  OFF_LR,  128, 512, 128, 512);

  compute_i2<<<24, 128, 0, stream>>>(cw3, cb3, cw4, cb4, i2all);
  compute_i2cond<<<24, 128, 0, stream>>>(i2all, cwo, cbo, i2cond);

  float* out_bbox = (float*)d_out;
  float* out_kl   = out_bbox + (size_t)BATCH * SEQL * 4;
  vae_main<<<BATCH / ROWS, 256, 0, stream>>>(
      label_set, bbox_input, eps, i2cond, wpk,
      cb1, cb2, cb5, pb1, pbm, pbv,
      eb1, eb2, eb3, ebm, ebv, db1, db2, db3, lb,
      out_bbox, out_kl);
}
